// SHDDelaysSNN_15831249453667
// MI455X (gfx1250) — matrix operand order repaired and checked by reading, not compiled, not run on hardware
//
#include <hip/hip_runtime.h>

// ---------------------------------------------------------------------------
// SHD delayed-SNN forward scan for gfx1250 (MI455X).
// Persistent cooperative kernel, WMMA f32_16x16x32_f16 for the two GEMMs,
// neuron state held in WMMA accumulator layout in registers across all steps.
// Spike operand tile staged per-block via global_load_async_to_lds_b128.
// GEMM K-loops software-pipelined (next fragments in flight under each WMMA).
// ---------------------------------------------------------------------------

typedef __attribute__((ext_vector_type(16))) _Float16 v16h;
typedef __attribute__((ext_vector_type(8)))  _Float16 v8h;
typedef __attribute__((ext_vector_type(8)))  float    v8f;

#define B_    128
#define T_    500
#define C_    700
#define H_    1024
#define O_    20
#define CPAD  704          // C padded to 22*32
#define KC    22           // K-chunks (of 32) over C
#define KH    32           // K-chunks over H
#define NT    64           // 16-wide N tiles over H
#define MT    8            // 16-wide M tiles over B
#define NBLK  64
#define NTHR  256

// workspace layout (bytes)
#define WS_CNT   0
#define WS_WIN   256
#define WIN_BYTES   ((size_t)NT * KC * 512 * 2)          // 1,441,792
#define WS_WREC  (WS_WIN + WIN_BYTES)
#define WREC_BYTES  ((size_t)NT * KH * 512 * 2)          // 2,097,152
#define WS_WOUT  (WS_WREC + WREC_BYTES)
#define WOUT_BYTES  ((size_t)2 * KH * 512 * 2)           // 65,536
#define WS_SPKA  (WS_WOUT + WOUT_BYTES)
#define SPKA_BYTES  ((size_t)MT * KH * 512 * 2)          // 262,144

// ---------------------------------------------------------------------------
// helpers
// ---------------------------------------------------------------------------

// A fragment (f16 16x32, M x K) from a row-major [16][32] half tile.
// Lane L holds M = L&15; halves 0..7 = K = (L>>4)*8 .. +7, halves 8..15 = +16.
__device__ __forceinline__ v16h load_afrag(const _Float16* tile, int lane) {
  int off = (lane & 15) * 32 + ((lane >> 4) << 3);
  union { v16h v; v8h h[2]; } u;
  u.h[0] = *(const v8h*)(tile + off);
  u.h[1] = *(const v8h*)(tile + off + 16);
  return u.v;
}

// B fragment (f16 32x16, K x N) from a row-major [32][16] half tile.
// Lane L holds N = L&15; halves 0..7 = K = (L>>4)*8 .. +7, halves 8..15 = +16.
__device__ __forceinline__ v16h load_bfrag(const _Float16* tile, int lane) {
  v16h r;
  const int n = lane & 15, kb = (lane >> 4) << 3;
#pragma unroll
  for (int s = 0; s < 16; ++s) {
    const int k = (s < 8) ? (kb + s) : (16 + kb + s - 8);
    r[s] = tile[k * 16 + n];
  }
  return r;
}

__device__ __forceinline__ void grid_sync(int* counter, int gen) {
  __syncthreads();
  if (threadIdx.x == 0) {
    __threadfence();  // publish this block's global writes device-wide
    __hip_atomic_fetch_add(counter, 1, __ATOMIC_RELEASE, __HIP_MEMORY_SCOPE_AGENT);
    const int target = gen * (int)gridDim.x;
    while (__hip_atomic_load(counter, __ATOMIC_ACQUIRE,
                             __HIP_MEMORY_SCOPE_AGENT) < target) {
      __builtin_amdgcn_s_sleep(1);
    }
  }
  __syncthreads();
}

// ---------------------------------------------------------------------------
// prep kernels
// ---------------------------------------------------------------------------

__global__ void init_ws_kernel(int* counter) {
  if (threadIdx.x == 0 && blockIdx.x == 0) *counter = 0;
}

// Repack a row-major [nrows][ncols] f32 weight matrix (out_features x in_features)
// into WMMA-B-fragment-friendly f16 tiles: dst[(nt*kchunks+ch)*512 + kk*16 + nn]
// = src[nt*16+nn][ch*32+kk], zero-padded.
__global__ void pack_wt_kernel(const float* __restrict__ src,
                               _Float16* __restrict__ dst,
                               int nrows, int ncols, int kchunks) {
  const int blk = blockIdx.x;
  const int nt = blk / kchunks;
  const int ch = blk - nt * kchunks;
  const long base = (long)blk * 512;
  for (int i = threadIdx.x; i < 512; i += blockDim.x) {
    const int kk = i >> 4;
    const int nn = i & 15;
    const int n = nt * 16 + nn;
    const int k = ch * 32 + kk;
    float v = 0.f;
    if (n < nrows && k < ncols) v = src[(long)n * ncols + k];
    dst[base + i] = (_Float16)v;
  }
}

// ---------------------------------------------------------------------------
// persistent scan kernel: 64 blocks x 256 threads (8 wave32 / block)
//   block -> m_tile = blk>>3 (batch rows b0..b0+15), n_group = blk&7
//   wave  -> n_tile = n_group*8 + wave  (H columns nt*16..+15)
// ---------------------------------------------------------------------------

__launch_bounds__(NTHR, 1)
__global__ void snn_scan_kernel(const float* __restrict__ x,
                                const int* __restrict__ delays,
                                const float* __restrict__ alpha,
                                const float* __restrict__ rho,
                                const float* __restrict__ beta_a,
                                const float* __restrict__ beta_out,
                                const _Float16* __restrict__ wInB,
                                const _Float16* __restrict__ wRecB,
                                const _Float16* __restrict__ wOutB,
                                _Float16* __restrict__ spkA,
                                int* __restrict__ counter,
                                float* __restrict__ out) {
  __shared__ __align__(32) _Float16 xt[KC * 512];       // delayed x tiles, 22 KB
  __shared__ __align__(32) _Float16 spkLds[KH * 512];   // spk_prev m-tile, 32 KB
  __shared__ int sdel[CPAD];

  const int tid  = threadIdx.x;
  const int lane = tid & 31;
  const int wv   = tid >> 5;
  const int blk  = blockIdx.x;
  const int mt   = blk >> 3;
  const int nt   = (blk & 7) * 8 + wv;
  const int b0   = mt * 16;
  const int lrow = lane & 15;
  const int lhi  = lane >> 4;

  // cache delays in LDS
  for (int i = tid; i < CPAD; i += NTHR) sdel[i] = (i < C_) ? delays[i] : 0;

  // zero initial spike buffer (spk_{t=-1} = 0), cooperatively across grid
  {
    const int tot = MT * KH * 512;
    for (int i = blk * NTHR + tid; i < tot; i += NBLK * NTHR)
      spkA[i] = (_Float16)0.f;
  }

  // per-lane neuron params for this wave's H column
  const int   hcol = nt * 16 + lrow;
  const float al   = alpha[hcol];
  const float rh   = rho[hcol];
  const float ba   = beta_a[hcol];

  // output waves: blocks with n_group==0, waves 0..1 cover O padded to 32
  const bool  is_out = ((blk & 7) == 0) && (wv < 2);
  const int   ot     = wv;                 // o-tile 0 or 1
  const int   ocol   = ot * 16 + lrow;
  const float bo     = (ocol < O_) ? beta_out[ocol] : 0.f;

  v8f v1   = {0.f, 0.f, 0.f, 0.f, 0.f, 0.f, 0.f, 0.f};
  v8f a1   = v1;
  v8f spk  = v1;
  v8f vout = v1;
  v8f osum = v1;

  int gen = 1;
  grid_sync(counter, gen);                  // spkA zeroed everywhere

  const _Float16* wb = wInB  + (long)nt * KC * 512;
  const _Float16* wr = wRecB + (long)nt * KH * 512;
  const _Float16* sa = spkA  + (long)mt * KH * 512;   // this block's spk m-tile
  const unsigned  spkLds0 = (unsigned)(size_t)&spkLds[0];

  for (int t = 0; t < T_; ++t) {
    // ---- Phase A0: async DMA spk_prev m-tile (32 KB) into LDS -------------
    // 2048 x 16B chunks over 256 threads: 8 global_load_async_to_lds_b128 each
#pragma unroll
    for (int j = 0; j < 8; ++j) {
      const int idx = j * NTHR + tid;                 // 16-byte chunk index
      asm volatile("global_load_async_to_lds_b128 %0, %1, off"
                   :: "v"(spkLds0 + idx * 16), "v"(sa + idx * 8)
                   : "memory");
    }

    // ---- Phase A1: gather delayed x rows for this m-tile into LDS ---------
    for (int i = tid; i < 16 * CPAD; i += NTHR) {
      const int r = i / CPAD;
      const int c = i - r * CPAD;
      float v = 0.f;
      if (c < C_) {
        const int tt = t - sdel[c];
        if (tt >= 0) v = x[((long)(b0 + r) * T_ + tt) * C_ + c];
      }
      xt[(c >> 5) * 512 + r * 32 + (c & 31)] = (_Float16)v;
    }
    asm volatile("s_wait_asynccnt 0x0" ::: "memory");   // DMA landed in LDS
    __syncthreads();

    // ---- Phase B: I1 tile = x_d @ w_in^T + spk_prev @ w_rec^T -------------
    // software-pipelined: next fragments in flight under each WMMA
    v8f acc = {0.f, 0.f, 0.f, 0.f, 0.f, 0.f, 0.f, 0.f};
    v16h a_cur = load_afrag(xt, lane);
    v16h b_cur = load_bfrag(wb, lane);
#pragma unroll 2
    for (int ch = 0; ch < KC; ++ch) {
      v16h a_nxt, b_nxt;
      if (ch + 1 < KC) {
        __builtin_prefetch(wb + (ch + 2) * 512, 0, 1);
        a_nxt = load_afrag(xt + (ch + 1) * 512, lane);
        b_nxt = load_bfrag(wb + (ch + 1) * 512, lane);
      } else {            // bridge the pipeline into the recurrent GEMM
        a_nxt = load_afrag(spkLds, lane);
        b_nxt = load_bfrag(wr, lane);
      }
      acc = __builtin_amdgcn_wmma_f32_16x16x32_f16(false, a_cur, false, b_cur,
                                                   (short)0, acc, false, false);
      a_cur = a_nxt; b_cur = b_nxt;
    }
#pragma unroll 2
    for (int ch = 0; ch < KH; ++ch) {
      v16h a_nxt = a_cur, b_nxt = b_cur;
      if (ch + 1 < KH) {
        __builtin_prefetch(wr + (ch + 2) * 512, 0, 1);
        a_nxt = load_afrag(spkLds + (ch + 1) * 512, lane);
        b_nxt = load_bfrag(wr + (ch + 1) * 512, lane);
      }
      acc = __builtin_amdgcn_wmma_f32_16x16x32_f16(false, a_cur, false, b_cur,
                                                   (short)0, acc, false, false);
      a_cur = a_nxt; b_cur = b_nxt;
    }

    ++gen;
    grid_sync(counter, gen);                // all spk_prev reads complete

    // ---- Phase C: adaptive-LIF update (state resident in registers) -------
#pragma unroll
    for (int r = 0; r < 8; ++r) {
      const float vn = al * v1[r] * (1.f - spk[r]) + (1.f - al) * (acc[r] - a1[r]);
      const float s  = (vn > 1.f) ? 1.f : 0.f;
      a1[r]  = rh * a1[r] + (1.f - rh) * (ba * vn + s);
      v1[r]  = vn;
      spk[r] = s;
    }
    // publish spk_n as f16 in A-fragment layout for next step / output GEMM
    {
      const int chs = nt >> 1;
      const int kk  = ((nt & 1) << 4) + lrow;       // h & 31
      _Float16* dst = spkA + ((long)mt * KH + chs) * 512 + kk;
#pragma unroll
      for (int r = 0; r < 8; ++r) {
        dst[(lhi * 8 + r) * 32] = (_Float16)spk[r];
      }
    }

    ++gen;
    grid_sync(counter, gen);                // spk_n visible device-wide

    // ---- Phase D: v_out update on output waves (reads NEW spikes) ---------
    if (is_out) {
      v8f ao = {0.f, 0.f, 0.f, 0.f, 0.f, 0.f, 0.f, 0.f};
      const _Float16* wo = wOutB + (long)ot * KH * 512;
      v16h a_c = load_afrag(sa, lane);
      v16h b_c = load_bfrag(wo, lane);
#pragma unroll 2
      for (int ch = 0; ch < KH; ++ch) {
        v16h a_n = a_c, b_n = b_c;
        if (ch + 1 < KH) {
          a_n = load_afrag(sa + (ch + 1) * 512, lane);
          b_n = load_bfrag(wo + (ch + 1) * 512, lane);
        }
        ao = __builtin_amdgcn_wmma_f32_16x16x32_f16(false, a_c, false, b_c,
                                                    (short)0, ao, false, false);
        a_c = a_n; b_c = b_n;
      }
#pragma unroll
      for (int r = 0; r < 8; ++r) {
        const float vn = bo * vout[r] + (1.f - bo) * ao[r];
        vout[r] = vn;
        osum[r] += vn;
      }
    }
  }

  // ---- finalize: out = out_sum / T  ([128, 20] f32) -------------------------
  if (is_out && ocol < O_) {
    const float inv_t = 1.f / (float)T_;
#pragma unroll
    for (int r = 0; r < 8; ++r) {
      const int b = b0 + lhi * 8 + r;
      out[(long)b * O_ + ocol] = osum[r] * inv_t;
    }
  }
}

// ---------------------------------------------------------------------------
// launch
// ---------------------------------------------------------------------------

extern "C" void kernel_launch(void* const* d_in, const int* in_sizes, int n_in,
                              void* d_out, int out_size, void* d_ws, size_t ws_size,
                              hipStream_t stream) {
  const float* x        = (const float*)d_in[0];   // [B,T,C]
  const int*   delays   = (const int*)  d_in[1];   // [C]
  const float* w_in     = (const float*)d_in[2];   // [H,C]
  const float* w_rec    = (const float*)d_in[3];   // [H,H]
  const float* w_out    = (const float*)d_in[4];   // [O,H]
  const float* alpha    = (const float*)d_in[5];   // [H]
  const float* rho      = (const float*)d_in[6];   // [H]
  const float* beta_a   = (const float*)d_in[7];   // [H]
  const float* beta_out = (const float*)d_in[8];   // [O]
  float*       out      = (float*)d_out;           // [B,O]

  char* ws = (char*)d_ws;
  int*      counter = (int*)      (ws + WS_CNT);
  _Float16* wInB    = (_Float16*) (ws + WS_WIN);
  _Float16* wRecB   = (_Float16*) (ws + WS_WREC);
  _Float16* wOutB   = (_Float16*) (ws + WS_WOUT);
  _Float16* spkA    = (_Float16*) (ws + WS_SPKA);

  init_ws_kernel<<<1, 64, 0, stream>>>(counter);
  pack_wt_kernel<<<NT * KC, 256, 0, stream>>>(w_in,  wInB,  H_, C_, KC);
  pack_wt_kernel<<<NT * KH, 256, 0, stream>>>(w_rec, wRecB, H_, H_, KH);
  pack_wt_kernel<<<2  * KH, 256, 0, stream>>>(w_out, wOutB, O_, H_, KH);

  snn_scan_kernel<<<NBLK, NTHR, 0, stream>>>(
      x, delays, alpha, rho, beta_a, beta_out,
      wInB, wRecB, wOutB, spkA, counter, out);
}
